// Head_68959994905100
// MI455X (gfx1250) — matrix operand order repaired and checked by reading, not compiled, not run on hardware
//
#include <hip/hip_runtime.h>

// Problem constants (from reference)
#define EMBED 768
#define HS    64
#define TSEQ  4096
#define BATCH 4

typedef __attribute__((ext_vector_type(16))) __bf16 v16bf;
typedef __attribute__((ext_vector_type(8)))  __bf16 v8bf;
typedef __attribute__((ext_vector_type(8)))  float  v8f;
typedef __attribute__((ext_vector_type(4)))  int    v4i;

static __device__ __forceinline__ v8f wmma_bf16(v16bf a, v16bf b, v8f c) {
  // D = A(16x32 bf16) x B(32x16 bf16) + C(16x16 f32)
  return __builtin_amdgcn_wmma_f32_16x16x32_bf16(false, a, false, b, (short)0, c,
                                                 false, false);
}

static __device__ __forceinline__ v16bf cat8(v8bf lo, v8bf hi) {
  v16bf r;
#pragma unroll
  for (int i = 0; i < 8; ++i) { r[i] = lo[i]; r[8 + i] = hi[i]; }
  return r;
}

// One 16-byte async global->LDS copy (GLOBAL_LOAD_ASYNC_TO_LDS_B128).
// Builtin signature (from hipcc diagnostic): arg0 is v4i* in addrspace(1).
static __device__ __forceinline__ void async_copy_b128(const void* g, void* l) {
  __builtin_amdgcn_global_load_async_to_lds_b128(
      (__attribute__((address_space(1))) v4i*)(v4i*)const_cast<void*>(g),
      (__attribute__((address_space(3))) v4i*)(v4i*)l, 0, 0);
}

// ---------------------------------------------------------------------------
// Kernel 0: W [E,H] fp32 -> Wt [H,E] bf16 (so WMMA B-operand loads are
// contiguous per lane).
// ---------------------------------------------------------------------------
__global__ __launch_bounds__(256) void head_wt_kernel(const float* __restrict__ W,
                                                      __bf16* __restrict__ Wt) {
  int i = blockIdx.x * 256 + threadIdx.x;
  if (i >= HS * EMBED) return;
  int h = i / EMBED;
  int e = i - h * EMBED;
  Wt[i] = (__bf16)W[e * HS + h];
}

// ---------------------------------------------------------------------------
// Kernel 1: QKV projection. 128 rows/block, 8 waves x 16 rows.
// A = x tile (16x32, fp32 -> bf16), B = Wt chunk (contiguous 32B/lane),
// 12 accumulators (Q,K,V x 4 n-tiles of 16).
// ---------------------------------------------------------------------------
__global__ __launch_bounds__(256) void head_qkv_kernel(
    const float* __restrict__ x,
    const __bf16* __restrict__ Wtq, const __bf16* __restrict__ Wtk,
    const __bf16* __restrict__ Wtv,
    __bf16* __restrict__ Q, __bf16* __restrict__ K, __bf16* __restrict__ V) {
  const int  lane = threadIdx.x & 31;
  const int  wave = threadIdx.x >> 5;
  const int  mrow = lane & 15;
  const int  hi   = (lane >> 4) & 1;
  const long row0 = (long)blockIdx.x * 128 + wave * 16;

  v8f aq[4] = {}, ak[4] = {}, av[4] = {};

  const float* xrow = x + (row0 + mrow) * EMBED;

  for (int kc = 0; kc < EMBED; kc += 32) {
    // A operand: four 16B chunks (16 fp32) per lane, fp32 -> bf16
    const float4* xv = (const float4*)(xrow + kc + (hi ? 16 : 0));
    float4 f0 = xv[0], f1 = xv[1], f2 = xv[2], f3 = xv[3];
    float  tf[16] = {f0.x, f0.y, f0.z, f0.w, f1.x, f1.y, f1.z, f1.w,
                     f2.x, f2.y, f2.z, f2.w, f3.x, f3.y, f3.z, f3.w};
    v16bf a;
#pragma unroll
    for (int j = 0; j < 16; ++j) a[j] = (__bf16)tf[j];

    const int bofs = kc + (hi ? 16 : 0);
#pragma unroll
    for (int nt = 0; nt < 4; ++nt) {
      const int h  = nt * 16 + mrow;
      v16bf     bq = *(const v16bf*)(Wtq + h * EMBED + bofs);
      v16bf     bk = *(const v16bf*)(Wtk + h * EMBED + bofs);
      v16bf     bv = *(const v16bf*)(Wtv + h * EMBED + bofs);
      aq[nt] = wmma_bf16(a, bq, aq[nt]);
      ak[nt] = wmma_bf16(a, bk, ak[nt]);
      av[nt] = wmma_bf16(a, bv, av[nt]);
    }
  }

  // C/D layout: elem r -> (m = r + 8*hi, n = mrow) within n-tile
#pragma unroll
  for (int nt = 0; nt < 4; ++nt) {
#pragma unroll
    for (int r = 0; r < 8; ++r) {
      const long row = row0 + r + (hi ? 8 : 0);
      const int  col = nt * 16 + mrow;
      Q[row * HS + col] = (__bf16)aq[nt][r];
      K[row * HS + col] = (__bf16)ak[nt][r];
      V[row * HS + col] = (__bf16)av[nt][r];
    }
  }
}

// ---------------------------------------------------------------------------
// Kernel 2: causal flash attention. 128 q-rows/block (8 waves x 16 rows),
// KV tile of 32.
//   * K and V tiles (32x64 bf16 row-major, 4 KB each) pulled into LDS once
//     per block with GLOBAL_LOAD_ASYNC_TO_LDS_B128 (one b128 per thread),
//     synchronized with s_wait_asynccnt + barrier.
//   * S = Q K^T via WMMA: B operand is a contiguous 32B ds read (row-major
//     storage already matches the B layout per lane).
//   * online softmax in f32; P transposed C->A layout through LDS.
//   * O += P V via WMMA with a pre-swizzled V tile built LDS->LDS.
// ---------------------------------------------------------------------------
__global__ __launch_bounds__(256) void head_attn_kernel(
    const __bf16* __restrict__ Q, const __bf16* __restrict__ K,
    const __bf16* __restrict__ V, float* __restrict__ out) {
  __shared__ __bf16         lk[32 * 64];    // K tile, row-major (4 KB)
  __shared__ __bf16         lraw[32 * 64];  // V tile, row-major (4 KB)
  __shared__ unsigned int   lv[4][32][8];   // V tile in WMMA-B layout (4 KB)
  __shared__ unsigned short lp[8][16][32];  // per-wave P tile 16x32 (8 KB)

  const int  lane  = threadIdx.x & 31;
  const int  wave  = threadIdx.x >> 5;
  const int  mrow  = lane & 15;
  const int  hi    = (lane >> 4) & 1;
  const int  batch = blockIdx.x >> 5;  // 32 q-blocks per batch
  const int  qb    = blockIdx.x & 31;
  const long base  = (long)batch * TSEQ;
  const int  qrow0 = qb * 128 + wave * 16;
  const float scale = 0.125f;  // 1/sqrt(64)

  // Q A-operand, resident for whole kernel (2 h-chunks of 32)
  v16bf qa[2];
#pragma unroll
  for (int c = 0; c < 2; ++c) {
    const __bf16* qr = Q + (base + qrow0 + mrow) * HS + c * 32 + (hi ? 16 : 0);
    qa[c] = cat8(*(const v8bf*)qr, *(const v8bf*)(qr + 8));
  }

  v8f   acco[4] = {};
  float mst[8], lst[8];
#pragma unroll
  for (int r = 0; r < 8; ++r) { mst[r] = -__builtin_inff(); lst[r] = 0.f; }

  // This thread's slice of the bulk tile copy: 16 bytes of row r at col c.
  const int cp_r = (int)threadIdx.x >> 3;       // 0..31
  const int cp_c = ((int)threadIdx.x & 7) * 8;  // 0..56 step 8

  const int send = qb * 128 + 128;  // causal bound for this block
  for (int s = 0; s < send; s += 32) {
    // ---- async bulk copy of K and V tiles (rows [s, s+32)) into LDS
    async_copy_b128(K + (base + s + cp_r) * HS + cp_c, lk + cp_r * HS + cp_c);
    async_copy_b128(V + (base + s + cp_r) * HS + cp_c, lraw + cp_r * HS + cp_c);
    if (s + 32 < send) {  // prefetch next K tile (global_prefetch_b8)
      __builtin_prefetch(K + (base + s + 32 + lane) * HS, 0, 1);
    }
    asm volatile("s_wait_asynccnt 0x0" ::: "memory");
    __syncthreads();

    // ---- swizzle V tile into WMMA-B layout (LDS -> regs -> LDS):
    // lv[nt][l][d] packs bf16 pair (k = 2d, 2d+1) at h = nt*16 + (l&15),
    // with k += 16 for lanes 16..31.
#pragma unroll
    for (int j = 0; j < 4; ++j) {
      const int flat = (int)threadIdx.x + 256 * j;
      const int d    = flat & 7;
      const int l    = (flat >> 3) & 31;
      const int nt   = flat >> 8;
      const int h    = nt * 16 + (l & 15);
      const int k0   = 2 * d + ((l & 16) ? 16 : 0);
      unsigned int u0 =
          (unsigned int)__builtin_bit_cast(unsigned short, lraw[k0 * HS + h]);
      unsigned int u1 = (unsigned int)__builtin_bit_cast(unsigned short,
                                                         lraw[(k0 + 1) * HS + h]);
      lv[nt][l][d] = u0 | (u1 << 16);
    }
    __syncthreads();

    if (s <= qrow0 + 15) {  // wave-uniform causal skip
      // ---- S = Q * K^T : B[k=h][n=srow] is a contiguous 32B LDS read
      v8f sacc[2];
#pragma unroll
      for (int st = 0; st < 2; ++st) {
        v8f c = {};
#pragma unroll
        for (int hc = 0; hc < 2; ++hc) {
          const __bf16* kr =
              lk + (st * 16 + mrow) * HS + hc * 32 + (hi ? 16 : 0);
          c = wmma_bf16(qa[hc], *(const v16bf*)kr, c);
        }
        sacc[st] = c;
      }

      // ---- mask + online softmax (rows live across 16-lane groups)
#pragma unroll
      for (int r = 0; r < 8; ++r) {
        const int qrow = qrow0 + r + (hi ? 8 : 0);
        float s0 = sacc[0][r] * scale;
        float s1 = sacc[1][r] * scale;
        if (s + mrow > qrow)      s0 = -__builtin_inff();
        if (s + 16 + mrow > qrow) s1 = -__builtin_inff();
        float mx = fmaxf(s0, s1);
#pragma unroll
        for (int msk = 1; msk <= 8; msk <<= 1)
          mx = fmaxf(mx, __shfl_xor(mx, msk, 32));
        const float mnew = fmaxf(mst[r], mx);
        const float corr = __expf(mst[r] - mnew);
        const float p0   = __expf(s0 - mnew);
        const float p1   = __expf(s1 - mnew);
        float rs = p0 + p1;
#pragma unroll
        for (int msk = 1; msk <= 8; msk <<= 1) rs += __shfl_xor(rs, msk, 32);
        lst[r] = lst[r] * corr + rs;
        mst[r] = mnew;
#pragma unroll
        for (int nt = 0; nt < 4; ++nt) acco[nt][r] *= corr;
        // P tile to LDS (C layout: row = r + 8*hi, col = st*16 + mrow)
        lp[wave][r + (hi ? 8 : 0)][mrow] =
            __builtin_bit_cast(unsigned short, (__bf16)p0);
        lp[wave][r + (hi ? 8 : 0)][16 + mrow] =
            __builtin_bit_cast(unsigned short, (__bf16)p1);
      }

      // cross-lane LDS dependency within the wave: wait for stores
      asm volatile("s_wait_dscnt 0" ::: "memory");

      // ---- reload P in A layout: lane<16 -> row mrow, K {0..15};
      // lane>=16 -> row mrow, K {16..31}
      const unsigned short* pr = &lp[wave][mrow][hi ? 16 : 0];
      v16bf pa = cat8(*(const v8bf*)pr, *(const v8bf*)(pr + 8));

      // ---- O += P * V
#pragma unroll
      for (int nt = 0; nt < 4; ++nt) {
        v16bf b  = *(const v16bf*)&lv[nt][lane][0];
        acco[nt] = wmma_bf16(pa, b, acco[nt]);
      }
    }
    __syncthreads();
  }

  // ---- epilogue: normalize by row sums, fp32 out [B,T,H]
#pragma unroll
  for (int nt = 0; nt < 4; ++nt) {
#pragma unroll
    for (int r = 0; r < 8; ++r) {
      const long row = base + qrow0 + r + (hi ? 8 : 0);
      out[row * HS + nt * 16 + mrow] = acco[nt][r] / lst[r];
    }
  }
}

// ---------------------------------------------------------------------------
extern "C" void kernel_launch(void* const* d_in, const int* in_sizes, int n_in,
                              void* d_out, int out_size, void* d_ws,
                              size_t ws_size, hipStream_t stream) {
  const float* x  = (const float*)d_in[0];
  const float* Wq = (const float*)d_in[1];
  const float* Wk = (const float*)d_in[2];
  const float* Wv = (const float*)d_in[3];
  float*       out = (float*)d_out;

  // Workspace layout (bytes):
  //   [0)       Wt_q 64x768 bf16 (98304) | [98304) Wt_k | [196608) Wt_v
  //   [294912)  Q bf16 [B*T,64] (2097152) | then K | then V  (~6.6 MB total)
  char*   ws  = (char*)d_ws;
  __bf16* Wtq = (__bf16*)(ws);
  __bf16* Wtk = Wtq + (size_t)HS * EMBED;
  __bf16* Wtv = Wtk + (size_t)HS * EMBED;
  __bf16* Qb  = (__bf16*)(ws + 3 * (size_t)HS * EMBED * 2);
  __bf16* Kb  = Qb + (size_t)BATCH * TSEQ * HS;
  __bf16* Vb  = Kb + (size_t)BATCH * TSEQ * HS;

  const int nW = HS * EMBED;  // 49152
  head_wt_kernel<<<(nW + 255) / 256, 256, 0, stream>>>(Wq, Wtq);
  head_wt_kernel<<<(nW + 255) / 256, 256, 0, stream>>>(Wk, Wtk);
  head_wt_kernel<<<(nW + 255) / 256, 256, 0, stream>>>(Wv, Wtv);

  const int rowBlocks = (BATCH * TSEQ) / 128;  // 128
  head_qkv_kernel<<<rowBlocks, 256, 0, stream>>>(x, Wtq, Wtk, Wtv, Qb, Kb, Vb);
  head_attn_kernel<<<rowBlocks, 256, 0, stream>>>(Qb, Kb, Vb, out);
}
